// GCNModule_27779848471368
// MI455X (gfx1250) — matrix operand order repaired and checked by reading, not compiled, not run on hardware
//
#include <hip/hip_runtime.h>

#define D 128
#define ROWITER 4            // row-tiles (of 64 rows) per GEMM block
#define EPW 4                // edges per wave in scatter

typedef __attribute__((ext_vector_type(16))) __bf16 v16bf;
typedef __attribute__((ext_vector_type(8)))  float  v8f;

// ---------------------------------------------------------------------------
// Zero-fill (float4 granularity)
// ---------------------------------------------------------------------------
__global__ void zero_f4(float4* __restrict__ p, int n4) {
  int i = blockIdx.x * blockDim.x + threadIdx.x;
  if (i < n4) p[i] = make_float4(0.f, 0.f, 0.f, 0.f);
}

// ---------------------------------------------------------------------------
// Edge scatter-add: one wave32 per EPW edges, lane handles 4 consecutive
// features. Feature table (51 MB) lives in the 192 MB L2, so the random
// gathers are L2 hits; we prefetch all EPW rows up front
// (global_prefetch_b8) to overlap the L2 latency, then gather + atomically
// accumulate (relaxed/agent -> global_atomic_add_f32).
// ---------------------------------------------------------------------------
__global__ void scatter_add_edges(const float* __restrict__ feat,
                                  const int* __restrict__ src,
                                  const int* __restrict__ dst,
                                  float* __restrict__ agg, int nedges) {
  const int lane = threadIdx.x & 31;
  const int wid  = (blockIdx.x * blockDim.x + threadIdx.x) >> 5;
  const int e0   = wid * EPW;
  if (e0 >= nedges) return;
  const int n = (nedges - e0 < EPW) ? (nedges - e0) : EPW;

  int sidx[EPW], didx[EPW];
  for (int i = 0; i < n; ++i) {
    sidx[i] = src[e0 + i];
    didx[i] = dst[e0 + i];
    __builtin_prefetch(feat + (size_t)sidx[i] * D + lane * 4, 0, 0);
  }
  for (int i = 0; i < n; ++i) {
    const float4 v = *reinterpret_cast<const float4*>(feat + (size_t)sidx[i] * D + lane * 4);
    float* o = agg + (size_t)didx[i] * D + lane * 4;
    __hip_atomic_fetch_add(o + 0, v.x, __ATOMIC_RELAXED, __HIP_MEMORY_SCOPE_AGENT);
    __hip_atomic_fetch_add(o + 1, v.y, __ATOMIC_RELAXED, __HIP_MEMORY_SCOPE_AGENT);
    __hip_atomic_fetch_add(o + 2, v.z, __ATOMIC_RELAXED, __HIP_MEMORY_SCOPE_AGENT);
    __hip_atomic_fetch_add(o + 3, v.w, __ATOMIC_RELAXED, __HIP_MEMORY_SCOPE_AGENT);
  }
}

// ---------------------------------------------------------------------------
// Fused LayerNorm + GEMM (+bias, +optional ReLU) via bf16 WMMA.
//   Y[row] = relu( LN(X[row]) @ W' + b' ),  W'[k][n] = g[k]*W[k][n],
//   b'[n] = b[n] + sum_k be[k]*W[k][n]   (gamma/beta folded into W/bias).
// Block = 128 threads = 4 waves. W' staged once per block into LDS
// (transposed, stride 136 bf16), then each wave loops over ROWITER 16-row
// tiles (block tile = 64*ROWITER rows) to amortize the W staging traffic.
// A: 16x32 bf16 fragments built in-register from LN'd row data.
// B: two 16B LDS reads per lane, 32B apart, per 32x16 bf16 fragment.
// ---------------------------------------------------------------------------
__global__ __launch_bounds__(128)
void ln_gemm_wmma(const float* __restrict__ X, const float* __restrict__ g,
                  const float* __restrict__ be, const float* __restrict__ W,
                  const float* __restrict__ bias, float* __restrict__ Y,
                  int nnodes, int do_relu) {
  constexpr int WPAD = 136;                // 128 + 8 bf16 pad (bank decorrelation)
  __shared__ __bf16 sW[D * WPAD];          // W' transposed: sW[n*WPAD + k]
  __shared__ float  sBias[D];

  const int tid = threadIdx.x;

  // Stage W' (gamma-folded, transposed) and fused bias. Global reads are
  // k-major (coalesced: n = tid, k = loop index).
  {
    float bacc = bias[tid];
    for (int k = 0; k < D; ++k) {
      float w = W[k * D + tid];
      sW[tid * WPAD + k] = (__bf16)(w * g[k]);
      bacc += be[k] * w;
    }
    sBias[tid] = bacc;
  }
  __syncthreads();

  const int lane = tid & 31;
  const int wave = tid >> 5;
  const int l16  = lane & 15;
  const int h    = lane >> 4;

  for (int it = 0; it < ROWITER; ++it) {
    const int rowbase = (blockIdx.x * ROWITER + it) * 64 + wave * 16;
    const int myrow   = rowbase + l16;                 // A-matrix row M = l16
    const int rc      = (myrow < nnodes) ? myrow : (nnodes - 1);

    // Lane owns 64 of its row's 128 elements: K = 16*c + 8*h + {0..7}, c=0..7.
    // (Lane L and L^16 own complementary halves of the same row.)
    float a[64];
    const float* xr = X + (size_t)rc * D + 8 * h;
#pragma unroll
    for (int c = 0; c < 8; ++c) {
      float4 u0 = *reinterpret_cast<const float4*>(xr + 16 * c);
      float4 u1 = *reinterpret_cast<const float4*>(xr + 16 * c + 4);
      a[8 * c + 0] = u0.x; a[8 * c + 1] = u0.y; a[8 * c + 2] = u0.z; a[8 * c + 3] = u0.w;
      a[8 * c + 4] = u1.x; a[8 * c + 5] = u1.y; a[8 * c + 6] = u1.z; a[8 * c + 7] = u1.w;
    }

    // Row statistics: local partial sums + exchange with partner lane (xor 16).
    float s = 0.f, s2 = 0.f;
#pragma unroll
    for (int i = 0; i < 64; ++i) { s += a[i]; s2 += a[i] * a[i]; }
    s  += __shfl_xor(s, 16);
    s2 += __shfl_xor(s2, 16);
    const float mu   = s * (1.0f / 128.0f);
    const float var  = s2 * (1.0f / 128.0f) - mu * mu;
    const float rsig = rsqrtf(var + 1e-5f);

    // A fragments: per ISA A-layout, frag[k][e] corresponds exactly to a[16k+e].
    v16bf fa[4];
#pragma unroll
    for (int k = 0; k < 4; ++k)
#pragma unroll
      for (int e = 0; e < 16; ++e)
        fa[k][e] = (__bf16)((a[16 * k + e] - mu) * rsig);

    // 8 N-tiles of 16 columns; K loop = 4 steps of 32.
#pragma unroll
    for (int nt = 0; nt < 8; ++nt) {
      v8f acc = {};
      const __bf16* wb = &sW[(size_t)(nt * 16 + l16) * WPAD + 8 * h];
#pragma unroll
      for (int k = 0; k < 4; ++k) {
        union { v16bf v; float4 f[2]; } ub;
        const float4* p = reinterpret_cast<const float4*>(wb + 32 * k);
        ub.f[0] = p[0];
        ub.f[1] = p[2];
        acc = __builtin_amdgcn_wmma_f32_16x16x32_bf16(
            /*neg_a=*/false, fa[k], /*neg_b=*/false, ub.v,
            /*c_mod=*/(short)0, acc, /*reuse_a=*/false, /*reuse_b=*/false);
      }
      // Epilogue: C/D layout — VGPR rr holds (M = 8*h + rr, N = l16).
      const int   col = nt * 16 + l16;
      const float bn  = sBias[col];
#pragma unroll
      for (int rr = 0; rr < 8; ++rr) {
        int   row = rowbase + 8 * h + rr;
        float v   = acc[rr] + bn;
        if (do_relu) v = fmaxf(v, 0.f);
        if (row < nnodes) Y[(size_t)row * D + col] = v;
      }
    }
  }
}

// ---------------------------------------------------------------------------
// Launch: 3 x (zero -> scatter-add -> LN+GEMM), ping-pong buffers in d_ws.
// ---------------------------------------------------------------------------
extern "C" void kernel_launch(void* const* d_in, const int* in_sizes, int n_in,
                              void* d_out, int out_size, void* d_ws, size_t ws_size,
                              hipStream_t stream) {
  const float* feat = (const float*)d_in[0];
  const int*   src  = (const int*)d_in[1];
  const int*   dst  = (const int*)d_in[2];
  const float* g1 = (const float*)d_in[3];
  const float* be1 = (const float*)d_in[4];
  const float* W1 = (const float*)d_in[5];
  const float* b1 = (const float*)d_in[6];
  const float* g2 = (const float*)d_in[7];
  const float* be2 = (const float*)d_in[8];
  const float* W2 = (const float*)d_in[9];
  const float* b2 = (const float*)d_in[10];
  const float* g3 = (const float*)d_in[11];
  const float* be3 = (const float*)d_in[12];
  const float* W3 = (const float*)d_in[13];
  const float* b3 = (const float*)d_in[14];

  const int nnodes = in_sizes[0] / D;
  const int nedges = in_sizes[1];

  float* bufA = (float*)d_ws;                       // aggregation buffer
  float* bufB = bufA + (size_t)nnodes * D;          // hidden activations

  const int n4 = (nnodes * D) / 4;
  dim3 tb256(256), tb128(128);
  dim3 zgrid((n4 + 255) / 256);
  const int swaves = (nedges + EPW - 1) / EPW;
  dim3 sgrid((swaves * 32 + 255) / 256);
  dim3 ggrid((nnodes + 64 * ROWITER - 1) / (64 * ROWITER));

  // Layer 1
  zero_f4<<<zgrid, tb256, 0, stream>>>((float4*)bufA, n4);
  scatter_add_edges<<<sgrid, tb256, 0, stream>>>(feat, src, dst, bufA, nedges);
  ln_gemm_wmma<<<ggrid, tb128, 0, stream>>>(bufA, g1, be1, W1, b1, bufB, nnodes, 1);

  // Layer 2
  zero_f4<<<zgrid, tb256, 0, stream>>>((float4*)bufA, n4);
  scatter_add_edges<<<sgrid, tb256, 0, stream>>>(bufB, src, dst, bufA, nedges);
  ln_gemm_wmma<<<ggrid, tb128, 0, stream>>>(bufA, g2, be2, W2, b2, bufB, nnodes, 1);

  // Layer 3 (no ReLU) -> d_out
  zero_f4<<<zgrid, tb256, 0, stream>>>((float4*)bufA, n4);
  scatter_add_edges<<<sgrid, tb256, 0, stream>>>(bufB, src, dst, bufA, nedges);
  ln_gemm_wmma<<<ggrid, tb128, 0, stream>>>(bufA, g3, be3, W3, b3, (float*)d_out, nnodes, 0);
}
